// ViT_27590869909604
// MI455X (gfx1250) — matrix operand order repaired and checked by reading, not compiled, not run on hardware
//
#include <hip/hip_runtime.h>
#include <cmath>

typedef __attribute__((ext_vector_type(16))) _Float16 v16h;
typedef __attribute__((ext_vector_type(8)))  _Float16 v8h;
typedef __attribute__((ext_vector_type(8)))  float    v8f;

// ---------------- problem dims ----------------
constexpr int Bsz   = 16;
constexpr int CHn   = 3;
constexpr int IMGS  = 384;
constexpr int PP    = 16;
constexpr int Dm    = 384;
constexpr int HEADS = 6;
constexpr int DHd   = 64;
constexpr int MLPD  = 1536;
constexpr int DEPTH = 6;
constexpr int INNER = HEADS * DHd;          // 384
constexpr int PDim  = PP * PP * 5 * CHn;    // 3840
constexpr int GR    = IMGS / PP;            // 24
constexpr int NPm   = GR * GR;              // 576
constexpr int SEQ   = NPm + 1;              // 577
constexpr int ROWS  = Bsz * SEQ;            // 9232 (divisible by 16)
constexpr int SEQP  = 608;                  // padded seq (19*32)

// ---------------- WMMA fragment loaders (CDNA5 wave32 layouts) ----------------
// A 16x32 f16 (MxK): lane L -> M = L%16 ; group g=L/16 ;
//   halves 0..7  = A[M][k0 + g*8 .. +7]
//   halves 8..15 = A[M][k0 + 16 + g*8 .. +7]
static __device__ __forceinline__
v16h load_a_frag(const _Float16* __restrict__ base, int ld, int row, int k0, int lane) {
    int g = lane >> 4;
    const _Float16* p = base + (size_t)row * ld + k0 + g * 8;
    union { v16h v; v8h h[2]; } u;
    u.h[0] = *(const v8h*)(p);
    u.h[1] = *(const v8h*)(p + 16);
    return u.v;
}
// B 32x16 f16 (KxN) from an N-major (transposed) panel Bt[n][k], ld = K:
//   lane L -> N = n0 + L%16 ; group g=L/16 ; halves 0..7 = Bt[N][k0 + g*8 .. +7], halves 8..15 = Bt[N][k0 + 16 + g*8 .. +7]
static __device__ __forceinline__
v16h load_b_frag(const _Float16* __restrict__ bt, int ld, int n0, int k0, int lane) {
    int n = lane & 15, g = lane >> 4;
    const _Float16* p = bt + (size_t)(n0 + n) * ld + k0 + g * 8;
    union { v16h v; v8h h[2]; } u;
    u.h[0] = *(const v8h*)(p);
    u.h[1] = *(const v8h*)(p + 16);
    return u.v;
}

// ---------------- CDNA5 async Global->LDS copy (ASYNCcnt path) ---------------
static __device__ __forceinline__
void async_to_lds_b128(unsigned lds_off, const _Float16* gaddr) {
    asm volatile("global_load_async_to_lds_b128 %0, %1, off"
                 :: "v"(lds_off), "v"(gaddr) : "memory");
}
static __device__ __forceinline__ void wait_asynccnt0() {
    asm volatile("s_wait_asynccnt 0" ::: "memory");
}

// ---------------- weight transpose + f32->f16 ----------------
__global__ void transpose_to_f16(const float* __restrict__ w, _Float16* __restrict__ wt,
                                 int K, int N) {
    long long i = (long long)blockIdx.x * blockDim.x + threadIdx.x;
    long long tot = (long long)K * N;
    if (i >= tot) return;
    int kk = (int)(i / N), nn = (int)(i % N);
    wt[(size_t)nn * K + kk] = (_Float16)w[i];
}

// ---------------- SPT gather + LayerNorm over PDim ----------------
__global__ void __launch_bounds__(256)
spt_patch_ln(const float* __restrict__ img, const float* __restrict__ g,
             const float* __restrict__ be, _Float16* __restrict__ out) {
    int rowid = blockIdx.x;                 // 0 .. Bsz*NPm-1
    int b = rowid / NPm, p = rowid % NPm;
    int ph = p / GR, pw = p % GR;
    int tid = threadIdx.x;
    __shared__ float buf[PDim];
    __shared__ float rs[256];
    __shared__ float rss[256];
    float s = 0.f, ss = 0.f;
    for (int e = tid; e < PDim; e += 256) {
        int c15 = e % 15; int t = e / 15;
        int p2 = t % PP;  int p1 = t / PP;
        int sh = c15 / 3, ch = c15 % 3;
        int dx = (sh == 1) ? -1 : (sh == 2) ? 1 : 0;
        int dy = (sh == 3) ? -1 : (sh == 4) ? 1 : 0;
        int y = ph * PP + p1 + dy;
        int x = pw * PP + p2 + dx;
        float v = 0.f;
        if (y >= 0 && y < IMGS && x >= 0 && x < IMGS)
            v = img[(((size_t)b * CHn + ch) * IMGS + y) * IMGS + x];
        buf[e] = v; s += v; ss += v * v;
    }
    rs[tid] = s; rss[tid] = ss; __syncthreads();
    for (int o = 128; o; o >>= 1) {
        if (tid < o) { rs[tid] += rs[tid + o]; rss[tid] += rss[tid + o]; }
        __syncthreads();
    }
    float mu = rs[0] / PDim;
    float var = rss[0] / PDim - mu * mu;
    float rstd = rsqrtf(var + 1e-5f);
    _Float16* orow = out + (size_t)rowid * PDim;
    for (int e = tid; e < PDim; e += 256)
        orow[e] = (_Float16)((buf[e] - mu) * rstd * g[e] + be[e]);
}

// ---------------- LayerNorm over D=384, wave per row ----------------
__global__ void __launch_bounds__(256)
ln_rows(const float* __restrict__ x, const float* __restrict__ g,
        const float* __restrict__ bta, _Float16* __restrict__ h) {
    int lane = threadIdx.x;
    int row = blockIdx.x * blockDim.y + threadIdx.y;
    if (row >= ROWS) return;
    const float* xr = x + (size_t)row * Dm;
    float s = 0.f, ss = 0.f;
    for (int c = lane; c < Dm; c += 32) { float v = xr[c]; s += v; ss += v * v; }
    for (int o = 16; o; o >>= 1) { s += __shfl_xor(s, o, 32); ss += __shfl_xor(ss, o, 32); }
    float mu = s / Dm;
    float var = ss / Dm - mu * mu;
    float r = rsqrtf(var + 1e-5f);
    _Float16* hr = h + (size_t)row * Dm;
    for (int c = lane; c < Dm; c += 32)
        hr[c] = (_Float16)((xr[c] - mu) * r * g[c] + bta[c]);
}

// ---------------- cls token + pos ----------------
__global__ void init_cls(const float* __restrict__ cls, const float* __restrict__ pos,
                         float* __restrict__ x) {
    int b = blockIdx.x; int d = threadIdx.x;   // 384 threads
    x[(size_t)b * SEQ * Dm + d] = cls[d] + pos[d];
}

// ------- 64x64 macro-tile WMMA GEMM, async double-buffered B panel in LDS ----
// Block = 128 thr (4 waves). Wave w computes rows [mt*16,mt*16+16) x 64 cols.
// B panel (64 N x 64 K f16) staged via GLOBAL_LOAD_ASYNC_TO_LDS_B128,
// double buffered; stage s+1 DMA overlaps WMMA compute of stage s.
// MODE 0: patch embed -> x (+spt_bias +pos, row remap b*SEQ+1+p)
// MODE 1: qkv -> q[BH][SEQP][64], k[BH][SEQP][64], vT[BH][64][SEQP]  (f16)
// MODE 2: out-proj -> x += A@Wo + bout      (residual)
// MODE 3: mlp1 -> u = gelu(A@W1 + b1)       (f16, ldo stride)
// MODE 4: mlp2 -> x += A@W2 + b2            (residual)
template <int MODE>
__global__ void __launch_bounds__(128)
gemm64(const _Float16* __restrict__ A, int lda,
       const _Float16* __restrict__ Bt, int ldb,
       int Mtiles, int Nblks, int K,
       const float* __restrict__ bias,
       const float* __restrict__ pos,
       float* __restrict__ xres,
       _Float16* __restrict__ oq,
       _Float16* __restrict__ ok,
       _Float16* __restrict__ ov,
       int ldo) {
    __shared__ __attribute__((aligned(16))) _Float16 bpanel[2][64 * 64]; // 2 x 8KB
    int lane = threadIdx.x & 31;
    int wave = threadIdx.x >> 5;
    int tid  = threadIdx.x;
    int nb = (int)(blockIdx.x % Nblks);
    int mb = (int)(blockIdx.x / Nblks);
    int mt_raw = mb * 4 + wave;
    int mt = (mt_raw < Mtiles) ? mt_raw : (Mtiles - 1);  // clamp tail, skip store
    int row = mt * 16 + (lane & 15);
    int n0 = nb * 64;

    // each thread DMAs 64B of the 8KB panel: row = tid/2, 32-half chunk = tid%2
    int prow = tid >> 1;
    int phalf = (tid & 1) * 32;
    unsigned lds_base0 = (unsigned)(size_t)(&bpanel[0][0]);
    unsigned lds_base1 = (unsigned)(size_t)(&bpanel[1][0]);
    auto stage = [&](int k0, int bufsel) {
        const _Float16* gp = Bt + (size_t)(n0 + prow) * ldb + k0 + phalf;
        unsigned lo = (bufsel ? lds_base1 : lds_base0) +
                      (unsigned)(prow * 64 + phalf) * 2u;
        async_to_lds_b128(lo, gp);
        async_to_lds_b128(lo + 32u, gp + 16);
    };

    v8f acc[4] = {};
    int nstages = K / 64;
    stage(0, 0);
    wait_asynccnt0();
    __syncthreads();
    for (int s = 0; s < nstages; ++s) {
        if (s + 1 < nstages) stage((s + 1) * 64, (s + 1) & 1);
        const _Float16* bp = &bpanel[s & 1][0];
#pragma unroll
        for (int kk = 0; kk < 64; kk += 32) {
            v16h a = load_a_frag(A, lda, row, s * 64 + kk, lane);
#pragma unroll
            for (int t = 0; t < 4; ++t) {     // 4 back-to-back WMMAs per A frag
                v16h b = load_b_frag(bp, 64, t * 16, kk, lane);
                acc[t] = __builtin_amdgcn_wmma_f32_16x16x32_f16(
                    false, a, false, b, (short)0, acc[t], false, false);
            }
        }
        wait_asynccnt0();   // stage s+1 DMA complete (overlapped with compute)
        __syncthreads();    // panel visible to all waves; buffer safe to reuse
    }

    if (mt_raw >= Mtiles) return;   // duplicate tail tile: discard
    int g = lane >> 4, n = lane & 15;
#pragma unroll
    for (int t = 0; t < 4; ++t) {
        int gc = n0 + t * 16 + n;
#pragma unroll
        for (int r = 0; r < 8; ++r) {
            int m = g * 8 + r;
            int gr = mt * 16 + m;
            float v = acc[t][r];
            if (MODE == 0) {                        // patch embed
                int b = gr / NPm, p = gr % NPm;
                int s = 1 + p;
                v += bias[gc] + pos[(size_t)s * Dm + gc];
                xres[((size_t)b * SEQ + s) * Dm + gc] = v;
            } else if (MODE == 1) {                 // qkv split + head layout
                int b = gr / SEQ, s = gr % SEQ;
                int part = gc / INNER;
                int ci = gc % INNER;
                int head = ci / DHd, dh = ci % DHd;
                int bh = b * HEADS + head;
                if (part == 0)
                    oq[((size_t)bh * SEQP + s) * DHd + dh] = (_Float16)v;
                else if (part == 1)
                    ok[((size_t)bh * SEQP + s) * DHd + dh] = (_Float16)v;
                else
                    ov[((size_t)bh * DHd + dh) * SEQP + s] = (_Float16)v;
            } else if (MODE == 2) {                 // out proj + residual
                size_t idx = (size_t)gr * Dm + gc;
                xres[idx] = xres[idx] + v + bias[gc];
            } else if (MODE == 3) {                 // mlp1 + exact gelu
                v += bias[gc];
                v = 0.5f * v * (1.f + erff(v * 0.70710678118654752f));
                oq[(size_t)gr * ldo + gc] = (_Float16)v;
            } else {                                // mlp2 + residual
                size_t idx = (size_t)gr * Dm + gc;
                xres[idx] = xres[idx] + v + bias[gc];
            }
        }
    }
}

// ---------------- fused LSA attention: per (b, head, 16-query tile) ----------
__global__ void __launch_bounds__(32)
attn_kernel(const _Float16* __restrict__ q, const _Float16* __restrict__ k,
            const _Float16* __restrict__ vT, const float* __restrict__ temp,
            int layer, _Float16* __restrict__ ao) {
    int lane = threadIdx.x;
    int qt = blockIdx.x;                    // 0..36
    int bh = blockIdx.y;                    // 0..95
    int b = bh / HEADS, head = bh % HEADS;
    int s0 = qt * 16;
    float scale = __expf(temp[layer]);
    __shared__ float    sc[16][SEQP];       // scores (f32)
    __shared__ _Float16 pr[16][SEQP];       // probs  (f16)
    const _Float16* qb = q  + (size_t)bh * SEQP * DHd;
    const _Float16* kb = k  + (size_t)bh * SEQP * DHd;
    const _Float16* vb = vT + (size_t)bh * DHd * SEQP;
    int g = lane >> 4, n = lane & 15;
    int m_row = s0 + (lane & 15);
    v16h a0 = load_a_frag(qb, DHd, m_row, 0, lane);
    v16h a1 = load_a_frag(qb, DHd, m_row, 32, lane);
    // ---- S = (Q Kt) * exp(t), diagonal + pad mask ----
    for (int j = 0; j < 37; ++j) {
        v16h b0 = load_b_frag(kb, DHd, j * 16, 0, lane);
        v16h b1 = load_b_frag(kb, DHd, j * 16, 32, lane);
        v8f c = {};
        c = __builtin_amdgcn_wmma_f32_16x16x32_f16(false, a0, false, b0, (short)0, c, false, false);
        c = __builtin_amdgcn_wmma_f32_16x16x32_f16(false, a1, false, b1, (short)0, c, false, false);
        int key = j * 16 + n;
        for (int r = 0; r < 8; ++r) {
            int m = g * 8 + r;
            int qi = s0 + m;
            float v = c[r] * scale;
            if (key >= SEQ || qi >= SEQ || qi == key) v = -3.0e38f;
            sc[m][key] = v;
        }
    }
    __syncthreads();
    // ---- row softmax (wave-cooperative) ----
    for (int m = 0; m < 16; ++m) {
        float mx = -3.0e38f;
        for (int c0 = lane; c0 < SEQP; c0 += 32) {
            float v = (c0 < 37 * 16) ? sc[m][c0] : -3.0e38f;
            mx = fmaxf(mx, v);
        }
        for (int o = 16; o; o >>= 1) mx = fmaxf(mx, __shfl_xor(mx, o, 32));
        float sum = 0.f;
        for (int c0 = lane; c0 < SEQP; c0 += 32) {
            float v = (c0 < 37 * 16) ? __expf(sc[m][c0] - mx) : 0.f;
            sc[m][c0] = v;
            sum += v;
        }
        for (int o = 16; o; o >>= 1) sum += __shfl_xor(sum, o, 32);
        float inv = 1.f / sum;
        for (int c0 = lane; c0 < SEQP; c0 += 32)
            pr[m][c0] = (_Float16)(sc[m][c0] * inv);
    }
    __syncthreads();
    // ---- O = P V  (4 column tiles of 16 over DH=64) ----
    const _Float16* prb = &pr[0][0];
    for (int t = 0; t < 4; ++t) {
        v8f o = {};
        for (int c0 = 0; c0 < SEQP; c0 += 32) {
            v16h pa = load_a_frag(prb, SEQP, lane & 15, c0, lane);
            v16h vbf = load_b_frag(vb, SEQP, t * 16, c0, lane);
            o = __builtin_amdgcn_wmma_f32_16x16x32_f16(false, pa, false, vbf,
                                                       (short)0, o, false, false);
        }
        for (int r = 0; r < 8; ++r) {
            int m = g * 8 + r;
            int s = s0 + m;
            if (s < SEQ)
                ao[((size_t)b * SEQ + s) * INNER + head * DHd + t * 16 + n] = (_Float16)o[r];
        }
    }
}

// ---------------- host orchestration ----------------
extern "C" void kernel_launch(void* const* d_in, const int* in_sizes, int n_in,
                              void* d_out, int out_size, void* d_ws, size_t ws_size,
                              hipStream_t stream) {
    (void)in_sizes; (void)n_in; (void)out_size; (void)ws_size;
    const float* img      = (const float*)d_in[0];
    const float* spt_g    = (const float*)d_in[1];
    const float* spt_b    = (const float*)d_in[2];
    const float* spt_w    = (const float*)d_in[3];
    const float* spt_bias = (const float*)d_in[4];
    const float* pos      = (const float*)d_in[5];
    const float* cls      = (const float*)d_in[6];
    const float* attn_g   = (const float*)d_in[7];
    const float* attn_b   = (const float*)d_in[8];
    const float* temp     = (const float*)d_in[9];
    const float* wqkv     = (const float*)d_in[10];
    const float* wout     = (const float*)d_in[11];
    const float* bout     = (const float*)d_in[12];
    const float* ff_g     = (const float*)d_in[13];
    const float* ff_b     = (const float*)d_in[14];
    const float* w1       = (const float*)d_in[15];
    const float* b1       = (const float*)d_in[16];
    const float* w2       = (const float*)d_in[17];
    const float* b2       = (const float*)d_in[18];

    char* ws = (char*)d_ws;
    size_t off = 0;
    auto alloc = [&](size_t bytes) {
        size_t r = off;
        off = (off + bytes + 255) & ~(size_t)255;
        return r;
    };
    _Float16* sptwT = (_Float16*)(ws + alloc((size_t)Dm * PDim * 2));
    _Float16* wqkvT = (_Float16*)(ws + alloc((size_t)DEPTH * 3 * INNER * Dm * 2));
    _Float16* woutT = (_Float16*)(ws + alloc((size_t)DEPTH * Dm * INNER * 2));
    _Float16* w1T   = (_Float16*)(ws + alloc((size_t)DEPTH * MLPD * Dm * 2));
    _Float16* w2T   = (_Float16*)(ws + alloc((size_t)DEPTH * Dm * MLPD * 2));
    // big region: patch activations; later reused for per-layer buffers
    size_t patch_bytes = (size_t)Bsz * NPm * PDim * 2;          // 70.8 MB
    char* region = ws + alloc(patch_bytes);
    _Float16* patchA = (_Float16*)region;
    // aliased sub-buffers (lifetimes disjoint from patchA)
    size_t ro = 0;
    auto sub = [&](size_t bytes) {
        size_t r = ro;
        ro = (ro + bytes + 255) & ~(size_t)255;
        return r;
    };
    _Float16* qbuf  = (_Float16*)(region + sub((size_t)Bsz * HEADS * SEQP * DHd * 2));
    _Float16* kbuf  = (_Float16*)(region + sub((size_t)Bsz * HEADS * SEQP * DHd * 2));
    _Float16* vTbuf = (_Float16*)(region + sub((size_t)Bsz * HEADS * DHd * SEQP * 2));
    _Float16* aobuf = (_Float16*)(region + sub((size_t)ROWS * INNER * 2));
    _Float16* ubuf  = (_Float16*)(region + sub((size_t)ROWS * MLPD * 2));
    float*    xbuf  = (float*)(ws + alloc((size_t)ROWS * Dm * 4));
    _Float16* hbuf  = (_Float16*)(ws + alloc((size_t)ROWS * Dm * 2));

    auto tlaunch = [&](const float* w, _Float16* wt, int K, int N) {
        long long tot = (long long)K * N;
        int blk = (int)((tot + 255) / 256);
        transpose_to_f16<<<blk, 256, 0, stream>>>(w, wt, K, N);
    };
    // weight prep (f32 -> f16, N-major)
    tlaunch(spt_w, sptwT, PDim, Dm);
    for (int l = 0; l < DEPTH; ++l) {
        tlaunch(wqkv + (size_t)l * Dm * 3 * INNER, wqkvT + (size_t)l * 3 * INNER * Dm, Dm, 3 * INNER);
        tlaunch(wout + (size_t)l * INNER * Dm,     woutT + (size_t)l * Dm * INNER,     INNER, Dm);
        tlaunch(w1   + (size_t)l * Dm * MLPD,      w1T   + (size_t)l * MLPD * Dm,      Dm, MLPD);
        tlaunch(w2   + (size_t)l * MLPD * Dm,      w2T   + (size_t)l * Dm * MLPD,      MLPD, Dm);
    }

    // SPT gather + LN -> patchA  (f16 [9216, 3840])
    spt_patch_ln<<<Bsz * NPm, 256, 0, stream>>>(img, spt_g, spt_b, patchA);

    auto ggrid = [&](int Mt, int Nb) {
        return dim3((unsigned)(((Mt + 3) / 4) * Nb));   // 4 m-tiles/block
    };
    // patch embed GEMM -> x (+bias +pos)
    gemm64<0><<<ggrid(Bsz * NPm / 16, Dm / 64), 128, 0, stream>>>(
        patchA, PDim, sptwT, PDim, Bsz * NPm / 16, Dm / 64, PDim,
        spt_bias, pos, xbuf, nullptr, nullptr, nullptr, 0);
    init_cls<<<Bsz, Dm, 0, stream>>>(cls, pos, xbuf);

    dim3 lnblk(32, 8);
    dim3 lngrd((ROWS + 7) / 8);
    for (int l = 0; l < DEPTH; ++l) {
        // h = LN(x; attn_g, attn_b)
        ln_rows<<<lngrd, lnblk, 0, stream>>>(xbuf, attn_g + (size_t)l * Dm,
                                             attn_b + (size_t)l * Dm, hbuf);
        // qkv
        gemm64<1><<<ggrid(ROWS / 16, 3 * INNER / 64), 128, 0, stream>>>(
            hbuf, Dm, wqkvT + (size_t)l * 3 * INNER * Dm, Dm,
            ROWS / 16, 3 * INNER / 64, Dm,
            nullptr, nullptr, nullptr, qbuf, kbuf, vTbuf, 0);
        // fused LSA attention
        attn_kernel<<<dim3(37, Bsz * HEADS), 32, 0, stream>>>(
            qbuf, kbuf, vTbuf, temp, l, aobuf);
        // out projection + residual
        gemm64<2><<<ggrid(ROWS / 16, Dm / 64), 128, 0, stream>>>(
            aobuf, INNER, woutT + (size_t)l * Dm * INNER, INNER,
            ROWS / 16, Dm / 64, INNER,
            bout + (size_t)l * Dm, nullptr, xbuf, nullptr, nullptr, nullptr, 0);
        // h = LN(x; ff_g, ff_b)
        ln_rows<<<lngrd, lnblk, 0, stream>>>(xbuf, ff_g + (size_t)l * Dm,
                                             ff_b + (size_t)l * Dm, hbuf);
        // mlp1 + gelu
        gemm64<3><<<ggrid(ROWS / 16, MLPD / 64), 128, 0, stream>>>(
            hbuf, Dm, w1T + (size_t)l * MLPD * Dm, Dm,
            ROWS / 16, MLPD / 64, Dm,
            b1 + (size_t)l * MLPD, nullptr, nullptr, ubuf, nullptr, nullptr, MLPD);
        // mlp2 + residual
        gemm64<4><<<ggrid(ROWS / 16, Dm / 64), 128, 0, stream>>>(
            ubuf, MLPD, w2T + (size_t)l * Dm * MLPD, MLPD,
            ROWS / 16, Dm / 64, MLPD,
            b2 + (size_t)l * Dm, nullptr, xbuf, nullptr, nullptr, nullptr, 0);
    }
    hipMemcpyAsync(d_out, xbuf, (size_t)ROWS * Dm * sizeof(float),
                   hipMemcpyDeviceToDevice, stream);
}
